// GraphAttentionModel_86535001079906
// MI455X (gfx1250) — matrix operand order repaired and checked by reading, not compiled, not run on hardware
//
#include <hip/hip_runtime.h>
#include <hip/hip_bf16.h>

typedef __attribute__((ext_vector_type(16))) __bf16 v16bf;
typedef __attribute__((ext_vector_type(8)))  float  v8f;

#define NEG_BIG (-9.0e15f)

// ---------------------------------------------------------------------------
// Transpose + f32->bf16 convert:  WT[h][o][k] = (o<O) ? W[h][k][o] : 0
// ---------------------------------------------------------------------------
__global__ void transpose_bf16_kernel(const float* __restrict__ W,
                                      __bf16* __restrict__ WT,
                                      int K, int O, int Opad,
                                      long w_hs, long wt_hs)
{
    int h = blockIdx.y;
    long idx = (long)blockIdx.x * blockDim.x + threadIdx.x;
    long total = (long)K * Opad;
    if (idx >= total) return;
    int k = (int)(idx % K);
    int o = (int)(idx / K);
    float v = (o < O) ? W[h * w_hs + (long)k * O + o] : 0.0f;
    WT[h * wt_hs + (long)o * K + k] = (__bf16)v;
}

// ---------------------------------------------------------------------------
// bf16 WMMA GEMM: D = A(f32,[M][K]) x B, with B pre-transposed bf16
// BT[Ncols][K].  One wave computes a 16 x (OT*16) strip, reusing the A
// fragment across OT col-tiles.  Also writes DT = D^T in bf16 for use as
// the attention V matrix.  grid = (M/64, 1, heads), block = 128 (4 waves).
// ---------------------------------------------------------------------------
template <int OT>
__global__ void gemm16_bf16_kernel(const float* __restrict__ A,
                                   const __bf16* __restrict__ BT,
                                   float* __restrict__ D,
                                   __bf16* __restrict__ DT,
                                   int M, int K, int ldD,
                                   long bt_hs, long d_hs, long dt_hs)
{
    int h = blockIdx.z;
    BT += h * bt_hs; D += h * d_hs; DT += h * dt_hs;

    int wave = threadIdx.x >> 5;
    int lane = threadIdx.x & 31;
    int row0 = blockIdx.x * 64 + wave * 16;
    int r  = lane & 15;          // row within tile (A) / col within tile (B,C)
    int hs = lane >> 4;          // half-select

    const float*  arow = A + (long)(row0 + r) * K;
    const __bf16* bcol = BT + (long)r * K + hs * 16;   // + t*16*K per col-tile

    v8f acc[OT];
#pragma unroll
    for (int t = 0; t < OT; ++t) acc[t] = (v8f){0, 0, 0, 0, 0, 0, 0, 0};

    for (int k0 = 0; k0 < K; k0 += 32) {
        // A fragment: elements 0..7 -> K = k0+hs*16+i ; 8..15 -> K = k0+hs*16+8+i
        v16bf a;
        const float* ap0 = arow + k0 + hs * 16;
        const float* ap1 = arow + k0 + hs * 16 + 8;
#pragma unroll
        for (int i = 0; i < 8; ++i) a[i]     = (__bf16)ap0[i];
#pragma unroll
        for (int i = 0; i < 8; ++i) a[8 + i] = (__bf16)ap1[i];
        __builtin_prefetch(ap0 + 128, 0, 0);           // 4 k-tiles ahead
#pragma unroll
        for (int t = 0; t < OT; ++t) {
            v16bf b = *(const v16bf*)(bcol + (long)t * 16 * K + k0);
            acc[t] = __builtin_amdgcn_wmma_f32_16x16x32_bf16(false, a, false, b,
                                                             (short)0, acc[t], false, false);
        }
    }
#pragma unroll
    for (int t = 0; t < OT; ++t) {
        int n = t * 16 + r;
#pragma unroll
        for (int i = 0; i < 8; ++i)
            D[(long)(row0 + hs * 8 + i) * ldD + n] = acc[t][i];
        __bf16* dt = DT + (long)n * M + row0 + hs * 8;
#pragma unroll
        for (int i = 0; i < 8; ++i) dt[i] = (__bf16)acc[t][i];
    }
}

// ---------------------------------------------------------------------------
// f1[h][n] = Wh[h][n][:] . a1[h][:]   (and same for f2)
// ---------------------------------------------------------------------------
__global__ void fvec_kernel(const float* __restrict__ Wh,
                            const float* __restrict__ a1,
                            const float* __restrict__ a2,
                            float* __restrict__ f1, float* __restrict__ f2,
                            int N, int O, int ldO, long wh_hs, long a_hs)
{
    int h = blockIdx.y;
    int n = blockIdx.x * blockDim.x + threadIdx.x;
    if (n >= N) return;
    const float* row = Wh + h * wh_hs + (long)n * ldO;
    const float* A1 = a1 + h * a_hs;
    const float* A2 = a2 + h * a_hs;
    float s1 = 0.f, s2 = 0.f;
    for (int o = 0; o < O; ++o) { float v = row[o]; s1 += v * A1[o]; s2 += v * A2[o]; }
    f1[(long)h * N + n] = s1;
    f2[(long)h * N + n] = s2;
}

// ---------------------------------------------------------------------------
// Online softmax statistics per (head,row): e = leakyrelu(f1[n]+f2[m]),
// masked by adj.  One wave32 per row; cross-lane (max,sum) merge.
// ---------------------------------------------------------------------------
__global__ void stats_kernel(const int* __restrict__ adj,
                             const float* __restrict__ f1,
                             const float* __restrict__ f2,
                             float* __restrict__ rowmax,
                             float* __restrict__ rowsum, int N)
{
    int h = blockIdx.y;
    int wave = threadIdx.x >> 5, lane = threadIdx.x & 31;
    int n = blockIdx.x * (blockDim.x >> 5) + wave;
    if (n >= N) return;
    float f1n = f1[(long)h * N + n];
    const int* arow = adj + (long)n * N;
    const float* F2 = f2 + (long)h * N;
    float mx = -3.0e38f, sum = 0.f;
    for (int m = lane; m < N; m += 32) {
        __builtin_prefetch(arow + m + 512, 0, 0);
        float e = f1n + F2[m];
        e = e > 0.f ? e : 0.2f * e;
        if (arow[m] == 0) e = NEG_BIG;
        if (e > mx) { sum = sum * __expf(mx - e) + 1.f; mx = e; }
        else        { sum += __expf(e - mx); }
    }
#pragma unroll
    for (int off = 16; off > 0; off >>= 1) {
        float omx  = __shfl_xor(mx, off, 32);
        float osum = __shfl_xor(sum, off, 32);
        float nmx = fmaxf(mx, omx);
        sum = sum * __expf(mx - nmx) + osum * __expf(omx - nmx);
        mx = nmx;
    }
    if (lane == 0) {
        rowmax[(long)h * N + n] = mx;
        rowsum[(long)h * N + n] = sum;
    }
}

// ---------------------------------------------------------------------------
// Fused attention: out[h][n][o] = sum_m P[n][m] * V[m][o] where
// P[n][m] = exp(mask(leakyrelu(f1[n]+f2[m])) - rowmax[n]) / rowsum[n].
// Each wave owns TWO 16-row A tiles (32 rows) x OT col-tiles so every V
// fragment load feeds 2 WMMAs.  V is pre-transposed bf16 VT[Ocols][N] so a
// B fragment is a single contiguous v16bf load.
// grid = (N/128, heads), block = 128 (4 waves, 32 rows each).
// ---------------------------------------------------------------------------
template <int OT>
__global__ void attn_kernel(const int* __restrict__ adj,
                            const float* __restrict__ f1,
                            const float* __restrict__ f2,
                            const float* __restrict__ rowmax,
                            const float* __restrict__ rowsum,
                            const __bf16* __restrict__ VT,
                            float* __restrict__ out,
                            int N, int ldO, long vt_hs, long out_hs)
{
    int h = blockIdx.y;
    int wave = threadIdx.x >> 5, lane = threadIdx.x & 31;
    int row0 = blockIdx.x * 128 + wave * 32;
    int r = lane & 15, hs = lane >> 4;
    long hN = (long)h * N;

    int rowA = row0 + r, rowB = row0 + 16 + r;
    float f1A = f1[hN + rowA],  f1B = f1[hN + rowB];
    float mxA = rowmax[hN + rowA], mxB = rowmax[hN + rowB];
    float riA = 1.0f / rowsum[hN + rowA], riB = 1.0f / rowsum[hN + rowB];
    const int*   arowA = adj + (long)rowA * N;
    const int*   arowB = adj + (long)rowB * N;
    const float* F2    = f2 + hN;
    const __bf16* vt   = VT + h * vt_hs + (long)r * N + hs * 16;

    v8f acc[2][OT];
#pragma unroll
    for (int s = 0; s < 2; ++s)
#pragma unroll
        for (int t = 0; t < OT; ++t) acc[s][t] = (v8f){0, 0, 0, 0, 0, 0, 0, 0};

    for (int m0 = 0; m0 < N; m0 += 32) {
        __builtin_prefetch(arowA + m0 + 128, 0, 0);    // 4 k-tiles ahead
        __builtin_prefetch(arowB + m0 + 128, 0, 0);
        __builtin_prefetch(F2 + m0 + 128, 0, 0);

        v16bf aA, aB;
        int kb = m0 + hs * 16;
#pragma unroll
        for (int half = 0; half < 2; ++half) {
            int base = kb + half * 8;
#pragma unroll
            for (int i = 0; i < 8; ++i) {
                int m = base + i;
                float f2m = F2[m];
                float eA = f1A + f2m;  eA = eA > 0.f ? eA : 0.2f * eA;
                float eB = f1B + f2m;  eB = eB > 0.f ? eB : 0.2f * eB;
                if (arowA[m] == 0) eA = NEG_BIG;
                if (arowB[m] == 0) eB = NEG_BIG;
                aA[half * 8 + i] = (__bf16)(__expf(eA - mxA) * riA);
                aB[half * 8 + i] = (__bf16)(__expf(eB - mxB) * riB);
            }
        }
#pragma unroll
        for (int t = 0; t < OT; ++t) {
            const __bf16* bp = vt + (long)t * 16 * N + m0;
            __builtin_prefetch(bp + 128, 0, 0);
            v16bf b = *(const v16bf*)bp;
            acc[0][t] = __builtin_amdgcn_wmma_f32_16x16x32_bf16(false, aA, false, b,
                                                                (short)0, acc[0][t], false, false);
            acc[1][t] = __builtin_amdgcn_wmma_f32_16x16x32_bf16(false, aB, false, b,
                                                                (short)0, acc[1][t], false, false);
        }
    }
#pragma unroll
    for (int s = 0; s < 2; ++s)
#pragma unroll
        for (int t = 0; t < OT; ++t) {
            int col = t * 16 + r;
#pragma unroll
            for (int i = 0; i < 8; ++i)
                out[h * out_hs + (long)(row0 + s * 16 + hs * 8 + i) * ldO + col] = acc[s][t][i];
        }
}

// ---------------------------------------------------------------------------
// hcat[n][h*O+o] = elu(attnout[h][n][o])    (f32, row-major [N][H*O])
// ---------------------------------------------------------------------------
__global__ void elu_cat_kernel(const float* __restrict__ attnout,
                               float* __restrict__ hcat, int N, int H, int O)
{
    int idx = blockIdx.x * blockDim.x + threadIdx.x;
    int total = N * H * O;
    if (idx >= total) return;
    int n = idx / (H * O);
    int j = idx % (H * O);
    int h = j / O, o = j % O;
    float v = attnout[((long)h * N + n) * O + o];
    hcat[idx] = v > 0.f ? v : __expf(v) - 1.f;
}

// ---------------------------------------------------------------------------
// log_softmax over C classes
// ---------------------------------------------------------------------------
__global__ void logsoftmax_kernel(const float* __restrict__ in,
                                  float* __restrict__ out, int N, int C, int ld)
{
    int n = blockIdx.x * blockDim.x + threadIdx.x;
    if (n >= N) return;
    const float* row = in + (long)n * ld;
    float mx = row[0];
    for (int c = 1; c < C; ++c) mx = fmaxf(mx, row[c]);
    float s = 0.f;
    for (int c = 0; c < C; ++c) s += __expf(row[c] - mx);
    float ls = __logf(s) + mx;
    for (int c = 0; c < C; ++c) out[(long)n * C + c] = row[c] - ls;
}

// ---------------------------------------------------------------------------
extern "C" void kernel_launch(void* const* d_in, const int* in_sizes, int n_in,
                              void* d_out, int out_size, void* d_ws, size_t ws_size,
                              hipStream_t stream)
{
    const int N = 4096, IN = 512, O = 64, H = 8, C = 10, CP = 16;

    const float* x   = (const float*)d_in[0];
    const int*   adj = (const int*)  d_in[1];
    const float* W   = (const float*)d_in[2];
    const float* a1  = (const float*)d_in[3];
    const float* a2  = (const float*)d_in[4];
    const float* Wo  = (const float*)d_in[5];
    const float* ao1 = (const float*)d_in[6];
    const float* ao2 = (const float*)d_in[7];
    float* dout = (float*)d_out;

    // workspace carve-up (aligned to 256B)
    char* p = (char*)d_ws;
    auto alloc = [&](size_t bytes) -> char* {
        char* q = p; p += (bytes + 255) & ~(size_t)255; return q;
    };
    __bf16* WT     = (__bf16*)alloc((size_t)H * O * IN * 2);     // [H][64][512]
    __bf16* WoT    = (__bf16*)alloc((size_t)CP * IN * 2);        // [16][512]
    float*  Whf    = (float*) alloc((size_t)H * N * O * 4);      // [H][N][64]
    __bf16* WhbT   = (__bf16*)alloc((size_t)H * O * N * 2);      // [H][64][N]
    float*  f1     = (float*) alloc((size_t)H * N * 4);
    float*  f2     = (float*) alloc((size_t)H * N * 4);
    float*  rmax   = (float*) alloc((size_t)H * N * 4);
    float*  rsum   = (float*) alloc((size_t)H * N * 4);
    float*  attnout= (float*) alloc((size_t)H * N * O * 4);      // [H][N][64]
    float*  hcat   = (float*) alloc((size_t)N * H * O * 4);      // [N][512]
    float*  Whqf   = (float*) alloc((size_t)N * CP * 4);         // [N][16]
    __bf16* WhqT   = (__bf16*)alloc((size_t)CP * N * 2);         // [16][N]
    float*  g1     = (float*) alloc((size_t)N * 4);
    float*  g2     = (float*) alloc((size_t)N * 4);
    float*  rmax2  = (float*) alloc((size_t)N * 4);
    float*  rsum2  = (float*) alloc((size_t)N * 4);
    float*  out2   = (float*) alloc((size_t)N * CP * 4);         // [N][16]

    // 1) transpose weights to bf16 [cols][K]
    transpose_bf16_kernel<<<dim3((IN * O + 255) / 256, H), 256, 0, stream>>>(
        W, WT, IN, O, O, (long)IN * O, (long)O * IN);
    transpose_bf16_kernel<<<dim3((IN * CP + 255) / 256, 1), 256, 0, stream>>>(
        Wo, WoT, IN, C, CP, 0, 0);

    // 2) Wh = x @ W[h]  (WMMA bf16), also emit WhbT for attention V
    gemm16_bf16_kernel<4><<<dim3(N / 64, 1, H), 128, 0, stream>>>(
        x, WT, Whf, WhbT, N, IN, O,
        (long)O * IN, (long)N * O, (long)O * N);

    // 3) f1/f2 attention projections
    fvec_kernel<<<dim3(N / 256, H), 256, 0, stream>>>(
        Whf, a1, a2, f1, f2, N, O, O, (long)N * O, (long)O);

    // 4) per-row softmax stats (online max/sum over masked rank-1 scores)
    stats_kernel<<<dim3(N / 8, H), 256, 0, stream>>>(adj, f1, f2, rmax, rsum, N);

    // 5) fused attention P @ Wh  (WMMA bf16)
    attn_kernel<4><<<dim3(N / 128, H), 128, 0, stream>>>(
        adj, f1, f2, rmax, rsum, WhbT, attnout, N, O, (long)O * N, (long)N * O);

    // 6) ELU + concat heads
    elu_cat_kernel<<<dim3((N * H * O + 255) / 256), 256, 0, stream>>>(
        attnout, hcat, N, H, O);

    // 7) output-layer GEMM: Whq = hcat @ Wo  (cols padded to 16)
    gemm16_bf16_kernel<1><<<dim3(N / 64, 1, 1), 128, 0, stream>>>(
        hcat, WoT, Whqf, WhqT, N, IN, CP, 0, 0, 0);

    // 8) g1/g2, stats, attention for output layer (single head)
    fvec_kernel<<<dim3(N / 256, 1), 256, 0, stream>>>(
        Whqf, ao1, ao2, g1, g2, N, C, CP, 0, 0);
    stats_kernel<<<dim3(N / 8, 1), 256, 0, stream>>>(adj, g1, g2, rmax2, rsum2, N);
    attn_kernel<1><<<dim3(N / 128, 1), 128, 0, stream>>>(
        adj, g1, g2, rmax2, rsum2, WhqT, out2, N, CP, 0, 0);

    // 9) log_softmax -> d_out [N][10]
    logsoftmax_kernel<<<dim3(N / 256), 256, 0, stream>>>(out2, dout, N, C, CP);
}
